// SimplePointNet_31112743092308
// MI455X (gfx1250) — matrix operand order repaired and checked by reading, not compiled, not run on hardware
//
#include <hip/hip_runtime.h>
#include <hip/hip_bf16.h>
#include <math.h>

// ---------------------------------------------------------------------------
// SimplePointNet on MI455X (gfx1250, wave32, WMMA)
//
// Both PointNet conv layers run entirely on v_wmma_f32_16x16x32_f16:
//  - layer 1: K padded 6(+bias)->32, one k-step, 4 N-tiles of 16 channels
//  - layer 2: K padded 67(+bias)->96, three k-steps, 8 N-tiles
// The geometric tail [rel(3), 1] rides in the A tile; W-tail rows + bias ride
// in the packed B, so the epilogue is pure atomic-max scatter.
// segment_max via int-punned atomicMax on zero-initialized accumulators
// (exact because relu(max(-inf,...)) == relu(max(0,...))), deterministic.
// Atomic pointers are hoisted per tile so the N-tile offset (nt*64B) folds
// into the instruction immediate.
// ---------------------------------------------------------------------------

typedef _Float16 half8_t  __attribute__((ext_vector_type(8)));
typedef _Float16 half16_t __attribute__((ext_vector_type(16)));
typedef _Float16 half4_t  __attribute__((ext_vector_type(4)));
typedef float    f32x8_t  __attribute__((ext_vector_type(8)));

#define HID1 64
#define HID2 128

#define B1_ELEMS (4 * 32 * 16)        // [nt(4)][lane(32)][j(16)]
#define B2_ELEMS (3 * 8 * 32 * 16)    // [kt(3)][nt(8)][lane(32)][j(16)]

// -------------------------------------------------------------- zero init
__global__ void zero_f4_kernel(float4* __restrict__ p, long n4) {
  long i  = (long)blockIdx.x * blockDim.x + threadIdx.x;
  long st = (long)gridDim.x * blockDim.x;
  for (; i < n4; i += st) p[i] = make_float4(0.f, 0.f, 0.f, 0.f);
}

// --------------------------------------- pack W1/b1/W2/b2 into f16 B layout
// B layout used here (same k order as A): lanes 0-15 hold K=32*kt+0..15 (j=K&15),
// lanes 16-31 hold K=32*kt+16..31.  Flat index bits: [kt][nt][lane(5)][j(4)].
__global__ void pack_w_kernel(const float* __restrict__ W1,
                              const float* __restrict__ b1,
                              const float* __restrict__ W2,
                              const float* __restrict__ b2,
                              _Float16* __restrict__ B1pack,
                              _Float16* __restrict__ B2pack) {
  int tid = blockIdx.x * blockDim.x + threadIdx.x;
  if (tid < B1_ELEMS) {
    // layer 1: single k-step (K=32), 4 N-tiles of 16 -> HID1=64
    int j    = tid & 15;
    int lane = (tid >> 4) & 31;
    int nt   = (tid >> 9) & 3;
    int g    = lane >> 4;
    int K    = 16 * g + j;                      // 0..31
    int n    = nt * 16 + (lane & 15);           // 0..63
    float v  = (K < 6) ? W1[K * HID1 + n] : (K == 6 ? b1[n] : 0.f);
    B1pack[tid] = (_Float16)v;
  } else if (tid < B1_ELEMS + B2_ELEMS) {
    // layer 2: 3 k-steps (K=96), 8 N-tiles of 16 -> HID2=128
    int t2   = tid - B1_ELEMS;
    int j    = t2 & 15;
    int lane = (t2 >> 4) & 31;
    int nt   = (t2 >> 9) & 7;
    int kt   = t2 >> 12;                        // 0..2
    int g    = lane >> 4;
    int K    = kt * 32 + 16 * g + j;            // 0..95
    int n    = nt * 16 + (lane & 15);           // 0..127
    float v  = (K < 67) ? W2[K * HID2 + n] : (K == 67 ? b2[n] : 0.f);
    B2pack[t2] = (_Float16)v;
  }
}

// ------------------------------------------------------------ layer 1 (WMMA)
// msg = [x[src](3), rel(3), 1] @ [W1; b1]  (K padded to 32), max into agg1
__global__ void __launch_bounds__(256)
layer1_wmma_kernel(const int* __restrict__ src, const int* __restrict__ dst,
                   const float* __restrict__ x,
                   const _Float16* __restrict__ B1pack,
                   int* __restrict__ agg1i, int E, int ntiles) {
  int wave = (int)((blockIdx.x * (unsigned)blockDim.x + threadIdx.x) >> 5);
  int lane = threadIdx.x & 31;
  if (wave >= ntiles) return;                 // wave-uniform: EXEC all-1
  int g = lane >> 4;
  int r = lane & 15;
  int gidx = min(wave * 16 + r, E - 1);       // clamp tail: max() idempotent
  int s = src[gidx], d = dst[gidx];
  float x0 = x[3 * s], x1 = x[3 * s + 1], x2 = x[3 * s + 2];
  float rx = x0 - x[3 * d], ry = x1 - x[3 * d + 1], rz = x2 - x[3 * d + 2];

  // A tile (16x32): lane group 0 holds K=0..7 in elems 0..7; rest zero.
  half16_t A = {};
  A[0] = (_Float16)(g ? 0.f : x0);
  A[1] = (_Float16)(g ? 0.f : x1);
  A[2] = (_Float16)(g ? 0.f : x2);
  A[3] = (_Float16)(g ? 0.f : rx);
  A[4] = (_Float16)(g ? 0.f : ry);
  A[5] = (_Float16)(g ? 0.f : rz);
  A[6] = (_Float16)(g ? 0.f : 1.f);           // bias lane (K=6)

  // hoisted atomic base pointers (one per owned C row)
  int* pm[8];
#pragma unroll
  for (int m = 0; m < 8; ++m) {
    int dm = __shfl(d, m + 8 * g);
    pm[m] = agg1i + (size_t)dm * HID1 + r;
  }

#pragma unroll
  for (int nt = 0; nt < 4; ++nt) {
    const _Float16* bp = B1pack + (size_t)(nt * 32 + lane) * 16;
    half8_t blo = *(const half8_t*)bp;
    half8_t bhi = *(const half8_t*)(bp + 8);
    half16_t B = __builtin_shufflevector(blo, bhi, 0, 1, 2, 3, 4, 5, 6, 7,
                                         8, 9, 10, 11, 12, 13, 14, 15);
    f32x8_t c = {0.f, 0.f, 0.f, 0.f, 0.f, 0.f, 0.f, 0.f};
    c = __builtin_amdgcn_wmma_f32_16x16x32_f16(false, A, false, B,
                                               (short)0, c, false, false);
#pragma unroll
    for (int m = 0; m < 8; ++m)
      atomicMax(pm[m] + nt * 16, __float_as_int((float)c[m]));
  }
}

// ------------------------------------------------- f32 -> f16 convert of h1
__global__ void cvt_f32_f16_kernel(const float4* __restrict__ in,
                                   half4_t* __restrict__ out, int n4) {
  int i = blockIdx.x * blockDim.x + threadIdx.x;
  if (i >= n4) return;
  float4 v = in[i];
  half4_t h;
  h.x = (_Float16)v.x; h.y = (_Float16)v.y;
  h.z = (_Float16)v.z; h.w = (_Float16)v.w;
  out[i] = h;
}

// ------------------------------------------------------------ layer 2 (WMMA)
// msg = [h1[src](64), rel(3), 1] @ [W2; b2]  (K padded to 96), max into agg2
__global__ void __launch_bounds__(256)
layer2_wmma_kernel(const int* __restrict__ src, const int* __restrict__ dst,
                   const float* __restrict__ x,
                   const _Float16* __restrict__ h1h,
                   const _Float16* __restrict__ B2pack,
                   int* __restrict__ agg2i, int E, int ntiles) {
  int wave = (int)((blockIdx.x * (unsigned)blockDim.x + threadIdx.x) >> 5);
  int lane = threadIdx.x & 31;
  if (wave >= ntiles) return;                 // wave-uniform: EXEC all-1
  int g = lane >> 4;
  int r = lane & 15;
  int gidx = min(wave * 16 + r, E - 1);       // clamp tail: max() idempotent
  int s = src[gidx], d = dst[gidx];
  float x0 = x[3 * s], x1 = x[3 * s + 1], x2 = x[3 * s + 2];
  float rx = x0 - x[3 * d], ry = x1 - x[3 * d + 1], rz = x2 - x[3 * d + 2];

  // A k-steps 0/1 (h1 part): per-lane contiguous 16B loads, K=16*g+0..15 as in B pack
  const _Float16* arow = h1h + (size_t)s * HID1;
  int o = 16 * g;
  half8_t a0lo = *(const half8_t*)(arow + o);
  half8_t a0hi = *(const half8_t*)(arow + o + 8);
  half8_t a1lo = *(const half8_t*)(arow + 32 + o);
  half8_t a1hi = *(const half8_t*)(arow + 32 + o + 8);
  half16_t A0 = __builtin_shufflevector(a0lo, a0hi, 0, 1, 2, 3, 4, 5, 6, 7,
                                        8, 9, 10, 11, 12, 13, 14, 15);
  half16_t A1 = __builtin_shufflevector(a1lo, a1hi, 0, 1, 2, 3, 4, 5, 6, 7,
                                        8, 9, 10, 11, 12, 13, 14, 15);
  // A k-step 2 (tail): K=64..67 -> {rel, 1} in lane group 0, elems 0..3.
  half16_t A2 = {};
  A2[0] = (_Float16)(g ? 0.f : rx);
  A2[1] = (_Float16)(g ? 0.f : ry);
  A2[2] = (_Float16)(g ? 0.f : rz);
  A2[3] = (_Float16)(g ? 0.f : 1.f);          // bias lane (K=67)

  // hoisted atomic base pointers
  int* pm[8];
#pragma unroll
  for (int m = 0; m < 8; ++m) {
    int dm = __shfl(d, m + 8 * g);
    pm[m] = agg2i + (size_t)dm * HID2 + r;
  }

#pragma unroll
  for (int nt = 0; nt < 8; ++nt) {
    const _Float16* bp0 = B2pack + (size_t)((0 * 8 + nt) * 32 + lane) * 16;
    const _Float16* bp1 = B2pack + (size_t)((1 * 8 + nt) * 32 + lane) * 16;
    const _Float16* bp2 = B2pack + (size_t)((2 * 8 + nt) * 32 + lane) * 16;
    half8_t b0lo = *(const half8_t*)bp0;
    half8_t b0hi = *(const half8_t*)(bp0 + 8);
    half8_t b1lo = *(const half8_t*)bp1;
    half8_t b1hi = *(const half8_t*)(bp1 + 8);
    half8_t b2lo = *(const half8_t*)bp2;
    half8_t b2hi = *(const half8_t*)(bp2 + 8);
    half16_t B0 = __builtin_shufflevector(b0lo, b0hi, 0, 1, 2, 3, 4, 5, 6, 7,
                                          8, 9, 10, 11, 12, 13, 14, 15);
    half16_t B1 = __builtin_shufflevector(b1lo, b1hi, 0, 1, 2, 3, 4, 5, 6, 7,
                                          8, 9, 10, 11, 12, 13, 14, 15);
    half16_t B2 = __builtin_shufflevector(b2lo, b2hi, 0, 1, 2, 3, 4, 5, 6, 7,
                                          8, 9, 10, 11, 12, 13, 14, 15);
    f32x8_t c = {0.f, 0.f, 0.f, 0.f, 0.f, 0.f, 0.f, 0.f};
    c = __builtin_amdgcn_wmma_f32_16x16x32_f16(false, A0, false, B0,
                                               (short)0, c, false, false);
    c = __builtin_amdgcn_wmma_f32_16x16x32_f16(false, A1, false, B1,
                                               (short)0, c, false, false);
    c = __builtin_amdgcn_wmma_f32_16x16x32_f16(false, A2, false, B2,
                                               (short)0, c, false, false);
#pragma unroll
    for (int m = 0; m < 8; ++m)
      atomicMax(pm[m] + nt * 16, __float_as_int((float)c[m]));
  }
}

// ------------------------------------------------- classifier + log_softmax
__global__ void __launch_bounds__(256)
classifier_kernel(const float* __restrict__ agg2, const float* __restrict__ Wc,
                  const float* __restrict__ bc, float* __restrict__ out, int N) {
  int nid = blockIdx.x * blockDim.x + threadIdx.x;
  if (nid >= N) return;
  const float* h = agg2 + (size_t)nid * HID2;   // already >= 0 (relu folded)
  float a0 = bc[0], a1 = bc[1], a2 = bc[2], a3 = bc[3], a4 = bc[4];
#pragma unroll 4
  for (int k = 0; k < HID2; ++k) {
    float hv = h[k];
    const float* w = Wc + k * 5;
    a0 = fmaf(hv, w[0], a0);
    a1 = fmaf(hv, w[1], a1);
    a2 = fmaf(hv, w[2], a2);
    a3 = fmaf(hv, w[3], a3);
    a4 = fmaf(hv, w[4], a4);
  }
  float m = fmaxf(fmaxf(fmaxf(a0, a1), fmaxf(a2, a3)), a4);
  float ssum = expf(a0 - m) + expf(a1 - m) + expf(a2 - m) +
               expf(a3 - m) + expf(a4 - m);
  float lse = logf(ssum) + m;
  float* o = out + (size_t)nid * 5;
  o[0] = a0 - lse; o[1] = a1 - lse; o[2] = a2 - lse;
  o[3] = a3 - lse; o[4] = a4 - lse;
}

// ---------------------------------------------------------------------------
extern "C" void kernel_launch(void* const* d_in, const int* in_sizes, int n_in,
                              void* d_out, int out_size, void* d_ws, size_t ws_size,
                              hipStream_t stream) {
  const float* x   = (const float*)d_in[0];
  const int*   ei  = (const int*)d_in[1];
  const float* W1  = (const float*)d_in[2];
  const float* b1  = (const float*)d_in[3];
  const float* W2  = (const float*)d_in[4];
  const float* b2  = (const float*)d_in[5];
  const float* Wc  = (const float*)d_in[6];
  const float* bc  = (const float*)d_in[7];

  const int N = in_sizes[0] / 3;
  const int E = in_sizes[1] / 2;
  const int* src = ei;
  const int* dst = ei + E;

  // workspace layout (sections stay 256B-aligned)
  char* ws = (char*)d_ws;
  size_t off = 0;
  float*    agg1   = (float*)(ws + off);    off += (size_t)N * HID1 * 4;  // 25.6 MB
  float*    agg2   = (float*)(ws + off);    off += (size_t)N * HID2 * 4;  // 51.2 MB
  _Float16* h1h    = (_Float16*)(ws + off); off += (size_t)N * HID1 * 2;  // 12.8 MB
  _Float16* B1pack = (_Float16*)(ws + off); off += B1_ELEMS * 2;          //  4 KB
  _Float16* B2pack = (_Float16*)(ws + off); off += B2_ELEMS * 2;          // 24 KB
  (void)ws_size; (void)n_in; (void)out_size;

  // 1) zero agg1+agg2 (contiguous): N*(64+128) floats
  long n4 = (long)N * (HID1 + HID2) / 4;
  zero_f4_kernel<<<4096, 256, 0, stream>>>((float4*)agg1, n4);

  // 2) pack W1/b1/W2/b2 into f16 WMMA-B layouts
  pack_w_kernel<<<(B1_ELEMS + B2_ELEMS + 255) / 256, 256, 0, stream>>>(
      W1, b1, W2, b2, B1pack, B2pack);

  int ntiles = (E + 15) / 16;
  int blocks = (ntiles + 7) / 8;  // 8 waves (256 threads) per block

  // 3) layer 1: WMMA (K=32) + atomic max
  layer1_wmma_kernel<<<blocks, 256, 0, stream>>>(src, dst, x, B1pack,
                                                 (int*)agg1, E, ntiles);

  // 4) h1 f32 -> f16
  int c4 = N * HID1 / 4;
  cvt_f32_f16_kernel<<<(c4 + 255) / 256, 256, 0, stream>>>(
      (const float4*)agg1, (half4_t*)h1h, c4);

  // 5) layer 2: WMMA (K=96, rel+bias folded into 3rd k-step) + atomic max
  layer2_wmma_kernel<<<blocks, 256, 0, stream>>>(src, dst, x, h1h, B2pack,
                                                 (int*)agg2, E, ntiles);

  // 6) classifier + log_softmax
  classifier_kernel<<<(N + 255) / 256, 256, 0, stream>>>(agg2, Wc, bc,
                                                         (float*)d_out, N);
}
